// SpatioTemporal_Convolution_38809324487397
// MI455X (gfx1250) — matrix operand order repaired and checked by reading, not compiled, not run on hardware
//
#include <hip/hip_runtime.h>

#define BB 16
#define TT 48
#define NN 1024
#define CIN 32
#define COUT 64

typedef __attribute__((ext_vector_type(16))) __bf16 v16bf;
typedef __attribute__((ext_vector_type(8)))  float  v8f;

union FragB16 { uint4 u[2]; v16bf v; };

__device__ __forceinline__ unsigned short f2bf(float f) {
  unsigned int u = __float_as_uint(f);
  u += 0x7FFFu + ((u >> 16) & 1u);
  return (unsigned short)(u >> 16);
}
__device__ __forceinline__ float bf2f(unsigned short h) {
  return __uint_as_float(((unsigned int)h) << 16);
}

// ---------------------------------------------------------------------------
// CDNA5 async global->LDS copy (ASYNCcnt-tracked), 16B per lane.
// Builtin signature (from hipcc diagnostic): (v4i AS1*, v4i AS3*, Imm, Imm)
// ---------------------------------------------------------------------------
#if __has_builtin(__builtin_amdgcn_global_load_async_to_lds_b128)
#define HAVE_ASYNC_LDS_BUILTIN 1
typedef int v4i128 __attribute__((vector_size(16)));
typedef __attribute__((address_space(1))) v4i128* gptr128;
typedef __attribute__((address_space(3))) v4i128* lptr128;
#endif
#if __has_builtin(__builtin_amdgcn_s_wait_asynccnt)
#define HAVE_WAIT_ASYNC_BUILTIN 1
#endif

__device__ __forceinline__ void async_cp16(const void* g, void* l) {
#if defined(__HIP_DEVICE_COMPILE__)
#ifdef HAVE_ASYNC_LDS_BUILTIN
  __builtin_amdgcn_global_load_async_to_lds_b128((gptr128)g, (lptr128)l, 0, 0);
#else
  asm volatile("global_load_async_to_lds_b128 %0, %1, off"
               :: "v"((unsigned int)(size_t)l), "v"(g)
               : "memory");
#endif
#endif
}
__device__ __forceinline__ void wait_async0() {
#if defined(__HIP_DEVICE_COMPILE__)
#ifdef HAVE_WAIT_ASYNC_BUILTIN
  __builtin_amdgcn_s_wait_asynccnt(0);
#else
  asm volatile("s_wait_asynccnt 0x0" ::: "memory");
#endif
#endif
}

// ---------------------------------------------------------------------------
// Kernel 1: rowsum + power iteration for lambda_max of L = diag(rowsum) - adj
// ---------------------------------------------------------------------------
__global__ void power_iter_kernel(const float* __restrict__ adj,
                                  float* __restrict__ rowsum_g,
                                  float* __restrict__ lam_g) {
  __shared__ float v[NN];
  __shared__ float red[NN];
  int i = threadIdx.x;
  const float* row = adj + (size_t)i * NN;
  float rs = 0.f;
  for (int j = 0; j < NN; ++j) rs += row[j];
  rowsum_g[i] = rs;
  v[i] = 0.03125f;
  __syncthreads();
  float lam = 1.f;
  for (int it = 0; it < 64; ++it) {
    float w = rs * v[i];
    for (int j = 0; j < NN; ++j) w -= row[j] * v[j];
    red[i] = w * w;
    __syncthreads();
    for (int s = NN / 2; s > 0; s >>= 1) {
      if (i < s) red[i] += red[i + s];
      __syncthreads();
    }
    lam = sqrtf(red[0]);
    __syncthreads();
    v[i] = w / lam;
    __syncthreads();
  }
  if (i == 0) lam_g[0] = lam;
}

// ---------------------------------------------------------------------------
// Kernel 2: L_tilde = 2*L/lambda - I
// ---------------------------------------------------------------------------
__global__ void ltilde_kernel(const float* __restrict__ adj,
                              const float* __restrict__ rowsum,
                              const float* __restrict__ lam_g,
                              float* __restrict__ Lt) {
  int idx = blockIdx.x * 256 + threadIdx.x;
  int i = idx >> 10, j = idx & 1023;
  float lam = lam_g[0];
  float L = ((i == j) ? rowsum[i] : 0.f) - adj[idx];
  Lt[idx] = 2.f * L / lam - ((i == j) ? 1.f : 0.f);
}

// ---------------------------------------------------------------------------
// Kernel 3: T2 = 2 * Lt @ Lt - I   (bf16 WMMA, f32 accumulate)
// 64x64 tile per workgroup; 8 waves x 2 tiles sharing one B fragment.
// ---------------------------------------------------------------------------
__global__ void __launch_bounds__(256)
t2_kernel(const float* __restrict__ Lt, float* __restrict__ T2) {
  __shared__ unsigned short As[64 * 32];  // [row64][k32]
  __shared__ unsigned short Bt[64 * 32];  // [col64][k32]
  int tid = threadIdx.x;
  int lane = tid & 31;
  int wv = tid >> 5;
  int l15 = lane & 15;
  int h = lane >> 4;
  int rt = wv >> 2;   // 0..1 (tiles rt and rt+2)
  int ctl = wv & 3;   // 0..3
  int row0 = blockIdx.y * 64;
  int col0 = blockIdx.x * 64;

  v8f acc0 = {0.f, 0.f, 0.f, 0.f, 0.f, 0.f, 0.f, 0.f};
  v8f acc1 = {0.f, 0.f, 0.f, 0.f, 0.f, 0.f, 0.f, 0.f};

  for (int kk = 0; kk < NN; kk += 32) {
    for (int e = tid; e < 2048; e += 256) {
      int r = e >> 5, c = e & 31;
      As[e] = f2bf(Lt[(size_t)(row0 + r) * NN + kk + c]);
      int kr = e >> 6, cl = e & 63;
      Bt[cl * 32 + kr] = f2bf(Lt[(size_t)(kk + kr) * NN + col0 + cl]);
    }
    __syncthreads();
    FragB16 a0, a1, bb;
    const char* ap = (const char*)As + (rt * 16 + l15) * 64 + h * 16;
    const uint4* pb = (const uint4*)((const char*)Bt + (ctl * 16 + l15) * 64 + h * 16);
    bb.u[0] = pb[0]; bb.u[1] = pb[2];
    const uint4* pa0 = (const uint4*)ap;
    a0.u[0] = pa0[0]; a0.u[1] = pa0[2];
    const uint4* pa1 = (const uint4*)(ap + 2048);
    a1.u[0] = pa1[0]; a1.u[1] = pa1[2];
    acc0 = __builtin_amdgcn_wmma_f32_16x16x32_bf16(false, a0.v, false, bb.v,
                                                   (short)0, acc0, false, false);
    acc1 = __builtin_amdgcn_wmma_f32_16x16x32_bf16(false, a1.v, false, bb.v,
                                                   (short)0, acc1, false, false);
    __syncthreads();
  }
  int col = col0 + ctl * 16 + l15;
#pragma unroll
  for (int i = 0; i < 8; ++i) {
    int m = i + 8 * h;
    int rg0 = row0 + rt * 16 + m;
    int rg1 = rg0 + 32;
    T2[(size_t)rg0 * NN + col] = 2.f * acc0[i] - ((rg0 == col) ? 1.f : 0.f);
    T2[(size_t)rg1 * NN + col] = 2.f * acc1[i] - ((rg1 == col) ? 1.f : 0.f);
  }
}

// ---------------------------------------------------------------------------
// Kernel 4: Mhat[b][k][i][j] = T_{k+1}[i][j]*att[b][i][j]  (bf16),
//           adiag[b][i] = att[b][i][i]
// ---------------------------------------------------------------------------
__global__ void pack_kernel(const float* __restrict__ att,
                            const float* __restrict__ Lt,
                            const float* __restrict__ T2,
                            unsigned short* __restrict__ Mhat,
                            float* __restrict__ adiag) {
  size_t idx = (size_t)blockIdx.x * 256 + threadIdx.x;  // B*N*N
  int ij = (int)(idx & (NN * NN - 1));
  int b = (int)(idx >> 20);
  int i = ij >> 10, j = ij & 1023;
  float a = att[idx];
  size_t base = ((size_t)b * 2) << 20;
  Mhat[base + ij] = f2bf(Lt[ij] * a);
  Mhat[base + (NN * NN) + ij] = f2bf(T2[ij] * a);
  if (i == j) adiag[b * NN + i] = a;
}

// ---------------------------------------------------------------------------
// Kernel 5: xbT[b][t][c][n] = (bf16) x[b][t][n][c]   (tiled transpose)
// ---------------------------------------------------------------------------
__global__ void xpose_kernel(const float* __restrict__ x,
                             unsigned short* __restrict__ xbT) {
  __shared__ float tile[32][33];
  int g = blockIdx.x;
  int ntile = g & 31;  // N/32 = 32 tiles
  int bt = g >> 5;     // b*T + t
  const float* src = x + (size_t)bt * NN * CIN + ntile * 32 * CIN;
  unsigned short* dst = xbT + (size_t)bt * CIN * NN + ntile * 32;
  int tid = threadIdx.x;
#pragma unroll
  for (int r = 0; r < 4; ++r) {
    int e = tid + r * 256;
    int nl = e >> 5, c = e & 31;
    tile[nl][c] = src[nl * CIN + c];
  }
  __syncthreads();
#pragma unroll
  for (int r = 0; r < 4; ++r) {
    int e = tid + r * 256;
    int c = e >> 5, nl = e & 31;
    dst[c * NN + nl] = f2bf(tile[nl][c]);
  }
}

// ---------------------------------------------------------------------------
// Main fused kernel: per (b, 32-row slab): chebyshev GCN (2 bf16 WMMA stages)
// + ReLU + temporal conv (3rd WMMA stage via LDS ring buffer) + bias + ReLU.
// x-tiles double-buffered via async global->LDS (ASYNCcnt).
// ---------------------------------------------------------------------------
#define AM_OFF   0         // 2 x 32 x 1024 bf16       = 131072
#define XS_OFF   131072    // 2 x (32 x 1024) bf16     = 131072 (double buffer)
#define RE_OFF   262144    // 32 x 96 bf16             = 6144
#define TH_OFF   268288    // 64 x 96 bf16             = 12288
#define GC_OFF   280576    // 3 x 32 x 64 bf16         = 12288
#define WC_OFF   292864    // 3 x 64 x 64 bf16         = 24576
#define AD_OFF   317440    // 32 f32                   = 128
#define SMEM_BYTES 317568

__device__ __forceinline__ void issue_xtile_async(const unsigned short* src,
                                                  char* dst, int tid) {
  for (int e = tid; e < 4096; e += 256)
    async_cp16((const char*)src + e * 16, dst + e * 16);
}

__global__ void __launch_bounds__(256)
stgcn_kernel(const unsigned short* __restrict__ Mhat,
             const unsigned short* __restrict__ xbT,
             const float* __restrict__ adiag,
             const float* __restrict__ theta,
             const float* __restrict__ convw,
             const float* __restrict__ convb,
             float* __restrict__ y) {
  extern __shared__ char smem[];
  unsigned short* aM   = (unsigned short*)(smem + AM_OFF);   // [sel][row32][k1024]
  unsigned short* rhsE = (unsigned short*)(smem + RE_OFF);   // [n32][cc96]
  unsigned short* ThT  = (unsigned short*)(smem + TH_OFF);   // [o64][cc96]
  unsigned short* gcnS = (unsigned short*)(smem + GC_OFF);   // [slot3][n32][co64]
  unsigned short* WcT  = (unsigned short*)(smem + WC_OFF);   // [dt3][cout64][cin64]
  float* adS           = (float*)(smem + AD_OFF);            // [32]

  int tid = threadIdx.x;
  int lane = tid & 31;
  int wv = tid >> 5;
  int b = blockIdx.y;
  int r0 = blockIdx.x * 32;

  // kick off async prefetch of x tile for t=0 before anything else
  issue_xtile_async(xbT + (size_t)(b * TT) * CIN * NN, smem + XS_OFF, tid);

  // ---- preload A slabs (reused over all T) ----
  {
    const uint4* s0 = (const uint4*)(Mhat + (((size_t)b * 2) << 20) + (size_t)r0 * NN);
    const uint4* s1 = (const uint4*)(Mhat + (((size_t)b * 2) << 20) + (1u << 20) + (size_t)r0 * NN);
    uint4* d0 = (uint4*)aM;
    uint4* d1 = (uint4*)(aM + 32 * NN);
    for (int e = tid; e < 4096; e += 256) { d0[e] = s0[e]; d1[e] = s1[e]; }
  }
  // ---- stacked theta, transposed for B-operand: ThT[o][cc] ----
  for (int e = tid; e < 64 * 96; e += 256) {
    int o = e / 96, cc = e % 96;
    float v;
    if (cc < 32)      v = theta[1 * CIN * COUT + cc * COUT + o];
    else if (cc < 64) v = theta[2 * CIN * COUT + (cc - 32) * COUT + o];
    else              v = theta[0 * CIN * COUT + (cc - 64) * COUT + o];
    ThT[e] = f2bf(v);
  }
  // ---- conv weights transposed: WcT[dt][cout][cin] = convw[dt][cin][cout] ----
  for (int e = tid; e < 3 * 64 * 64; e += 256) {
    int dt = e >> 12; int rem = e & 4095; int co = rem >> 6; int ci = rem & 63;
    WcT[(dt << 12) + (co << 6) + ci] = f2bf(convw[(dt << 12) + (ci << 6) + co]);
  }
  if (tid < 32) adS[tid] = adiag[b * NN + r0 + tid];
  __syncthreads();

  int rh = wv >> 2;          // row half of 32-row slab
  int ct = wv & 3;           // stage-1 tile column index
  int sel1 = ct >> 1;        // M1 or M2
  int c0 = (ct & 1) * 16;    // channel offset within CIN
  int o0 = ct * 16;          // output-channel tile for stage2/conv
  int l15 = lane & 15;
  int h = lane >> 4;
  float bias = convb[o0 + l15];

  for (int t = 0; t <= TT; ++t) {
    if (t < TT) {
      // ---- wait for this t's async x tile; prefetch t+1 into other buffer ----
      wait_async0();
      __syncthreads();
      if (t + 1 < TT)
        issue_xtile_async(xbT + (size_t)(b * TT + t + 1) * CIN * NN,
                          smem + XS_OFF + ((t + 1) & 1) * 65536, tid);
      const char* xsT = smem + XS_OFF + (t & 1) * 65536;  // [c32][n1024]

      // ---- stage 1: rhs_k tile = M_k(16x1024) @ x(1024x16)  K=1024 ----
      v8f acc = {0.f, 0.f, 0.f, 0.f, 0.f, 0.f, 0.f, 0.f};
      {
        const char* aBase = (const char*)aM + sel1 * 65536 + (rh * 16 + l15) * 2048 + h * 16;
        const char* bBase = xsT + (c0 + l15) * 2048 + h * 16;
        for (int kk = 0; kk < NN; kk += 32) {
          FragB16 a, bb;
          const uint4* pa = (const uint4*)(aBase + kk * 64);
          a.u[0] = pa[0]; a.u[1] = pa[2];
          const uint4* pb = (const uint4*)(bBase + kk * 64);
          bb.u[0] = pb[0]; bb.u[1] = pb[2];
          acc = __builtin_amdgcn_wmma_f32_16x16x32_bf16(false, a.v, false, bb.v,
                                                        (short)0, acc, false, false);
        }
      }
#pragma unroll
      for (int i = 0; i < 8; ++i) {
        int m = i + 8 * h;
        rhsE[(rh * 16 + m) * 96 + ct * 16 + l15] = f2bf(acc[i]);
      }
      // cols 64..95: diag(att) * x  (the T0 = I term)
      for (int e = tid; e < 32 * 32; e += 256) {
        int nl = e >> 5, c = e & 31;
        float v = adS[nl] * bf2f(*(const unsigned short*)(xsT + (c * NN + r0 + nl) * 2));
        rhsE[nl * 96 + 64 + c] = f2bf(v);
      }
      __syncthreads();

      // ---- stage 2: gcn = rhsE(32x96) @ Theta(96x64), ReLU -> ring buffer ----
      v8f g = {0.f, 0.f, 0.f, 0.f, 0.f, 0.f, 0.f, 0.f};
      {
        const char* a2 = (const char*)rhsE + (rh * 16 + l15) * 192 + h * 16;
        const char* b2 = (const char*)ThT + (o0 + l15) * 192 + h * 16;
#pragma unroll
        for (int q = 0; q < 3; ++q) {
          FragB16 a, bb;
          const uint4* pa = (const uint4*)(a2 + q * 64);
          a.u[0] = pa[0]; a.u[1] = pa[2];
          const uint4* pb = (const uint4*)(b2 + q * 64);
          bb.u[0] = pb[0]; bb.u[1] = pb[2];
          g = __builtin_amdgcn_wmma_f32_16x16x32_bf16(false, a.v, false, bb.v,
                                                      (short)0, g, false, false);
        }
      }
      unsigned short* gs = gcnS + (t % 3) * (32 * 64);
#pragma unroll
      for (int i = 0; i < 8; ++i) {
        int m = i + 8 * h;
        float v = g[i] > 0.f ? g[i] : 0.f;
        gs[(rh * 16 + m) * 64 + o0 + l15] = f2bf(v);
      }
    }
    __syncthreads();

    // ---- stage 3: temporal conv for t_out = t-1 (3 taps, K=64 each) ----
    if (t >= 1) {
      int to = t - 1;
      v8f ya = {0.f, 0.f, 0.f, 0.f, 0.f, 0.f, 0.f, 0.f};
      const char* bw = (const char*)WcT + (o0 + l15) * 128 + h * 16;
      for (int dt = -1; dt <= 1; ++dt) {
        int ts = to + dt;
        if (ts < 0 || ts >= TT) continue;   // uniform branch: EXEC stays full
        const char* ab = (const char*)gcnS + (ts % 3) * 4096 + (rh * 16 + l15) * 128 + h * 16;
#pragma unroll
        for (int q = 0; q < 2; ++q) {
          FragB16 a, bb;
          const uint4* pa = (const uint4*)(ab + q * 64);
          a.u[0] = pa[0]; a.u[1] = pa[2];
          const uint4* pb = (const uint4*)(bw + (dt + 1) * 8192 + q * 64);
          bb.u[0] = pb[0]; bb.u[1] = pb[2];
          ya = __builtin_amdgcn_wmma_f32_16x16x32_bf16(false, a.v, false, bb.v,
                                                       (short)0, ya, false, false);
        }
      }
      float* yo = y + ((size_t)(b * TT + to) * NN + r0 + rh * 16) * COUT + o0 + l15;
#pragma unroll
      for (int i = 0; i < 8; ++i) {
        int m = i + 8 * h;
        float v = ya[i] + bias;
        yo[(size_t)m * COUT] = v > 0.f ? v : 0.f;
      }
    }
    __syncthreads();
  }
}

// ---------------------------------------------------------------------------
// Workspace layout (bytes)
// ---------------------------------------------------------------------------
#define WS_ROWSUM 0
#define WS_LAM    4096
#define WS_LT     8192
#define WS_T2     4202496
#define WS_ADIAG  8396800
#define WS_MHAT   8462336
#define WS_XBT    75571200

extern "C" void kernel_launch(void* const* d_in, const int* in_sizes, int n_in,
                              void* d_out, int out_size, void* d_ws, size_t ws_size,
                              hipStream_t stream) {
  const float* x     = (const float*)d_in[0];
  const float* adj   = (const float*)d_in[1];
  const float* att   = (const float*)d_in[2];
  const float* theta = (const float*)d_in[3];
  const float* convw = (const float*)d_in[4];
  const float* convb = (const float*)d_in[5];
  float* y = (float*)d_out;
  char* ws = (char*)d_ws;

  float* rowsum = (float*)(ws + WS_ROWSUM);
  float* lam    = (float*)(ws + WS_LAM);
  float* Lt     = (float*)(ws + WS_LT);
  float* T2     = (float*)(ws + WS_T2);
  float* adiag  = (float*)(ws + WS_ADIAG);
  unsigned short* Mhat = (unsigned short*)(ws + WS_MHAT);
  unsigned short* xbT  = (unsigned short*)(ws + WS_XBT);

  power_iter_kernel<<<1, 1024, 0, stream>>>(adj, rowsum, lam);
  ltilde_kernel<<<4096, 256, 0, stream>>>(adj, rowsum, lam, Lt);
  t2_kernel<<<dim3(16, 16), 256, 0, stream>>>(Lt, T2);
  pack_kernel<<<65536, 256, 0, stream>>>(att, Lt, T2, Mhat, adiag);
  xpose_kernel<<<BB * TT * (NN / 32), 256, 0, stream>>>(x, xbT);

  (void)hipFuncSetAttribute((const void*)stgcn_kernel,
                            hipFuncAttributeMaxDynamicSharedMemorySize, SMEM_BYTES);
  stgcn_kernel<<<dim3(NN / 32, BB), 256, SMEM_BYTES, stream>>>(
      Mhat, xbT, adiag, theta, convw, convb, y);
}
